// CoAttention_77592879169836
// MI455X (gfx1250) — matrix operand order repaired and checked by reading, not compiled, not run on hardware
//
#include <hip/hip_runtime.h>

// ---------------------------------------------------------------------------
// CDNA5 (gfx1250) co-attention: all GEMMs via v_wmma_f32_16x16x32_bf16,
// global->LDS tile staging via the Tensor Data Mover where layout permits.
// B=8, L1=L2=2048, D=1024.
// ---------------------------------------------------------------------------

typedef __attribute__((ext_vector_type(16))) __bf16 v16bf;
typedef __attribute__((ext_vector_type(8)))  __bf16 v8bf;
typedef __attribute__((ext_vector_type(8)))  float  v8f;
typedef __attribute__((ext_vector_type(4)))  unsigned int u32x4;
typedef __attribute__((ext_vector_type(8)))  int i32x8;
typedef __attribute__((ext_vector_type(4)))  int i32x4;

#define BM 128
#define BN 128
#define BK 32
#define LDSA 40   // padded LDS row stride (bf16 elems): 64B data + 16B pad
#define LDSB 40

#if defined(__HIP_DEVICE_COMPILE__) && __has_builtin(__builtin_amdgcn_tensor_load_to_lds)
#define HAVE_TDM 1
#else
#define HAVE_TDM 0
#endif

#if HAVE_TDM
// ---------------------------------------------------------------------------
// TDM 2D tile load: tile_rows rows of tile_k bf16 elements, row stride
// row_stride_elems, packed into LDS with 16B padding after each 64B row
// (so LDS row stride = 80B = LDSA/LDSB bf16 elems).
// D# packing per CDNA5 ISA.
// ---------------------------------------------------------------------------
__device__ __forceinline__ void tdm_load_tile_bf16(unsigned lds_byte_addr,
                                                   const __bf16* gsrc,
                                                   int tile_k, int tile_rows,
                                                   long long row_stride_elems)
{
    const unsigned long long ga = (unsigned long long)(const void*)gsrc;

    u32x4 g0;
    g0.x = 1u;                                              // count=1 (valid)
    g0.y = lds_byte_addr;                                   // lds_addr
    g0.z = (unsigned)ga;                                    // global_addr lo
    g0.w = (unsigned)((ga >> 32) & 0x01ffffffull) | (2u << 30);  // hi | type=2

    const unsigned long long tdim0 = 1ull << 30;            // generous OOB dims
    const unsigned long long tdim1 = 1ull << 30;
    const unsigned long long st    = (unsigned long long)row_stride_elems;

    // group1 bits: [17:16] data_size=1(2B) [20] pad_en [24:22] pad_interval=3
    // (16 dwords) [31:25] pad_amount=3 (4 dwords) [79:48] tensor_dim0
    // [111:80] tensor_dim1 [127:112] tile_dim0 [143:128] tile_dim1
    // [207:160] tensor_dim0_stride
    unsigned long long q0 = (1ull << 16) | (1ull << 20) | (3ull << 22) |
                            (3ull << 25) | ((tdim0 & 0xffffull) << 48);
    unsigned long long q1 = (tdim0 >> 16) | ((tdim1 & 0xffffffffull) << 16) |
                            ((unsigned long long)(unsigned)tile_k << 48);
    unsigned long long q2 = (unsigned long long)(unsigned)tile_rows |
                            ((st & 0xffffffffull) << 32);
    unsigned long long q3 = (st >> 32) & 0xffffull;

    i32x8 g1;
    g1[0] = (int)(unsigned)q0;  g1[1] = (int)(unsigned)(q0 >> 32);
    g1[2] = (int)(unsigned)q1;  g1[3] = (int)(unsigned)(q1 >> 32);
    g1[4] = (int)(unsigned)q2;  g1[5] = (int)(unsigned)(q2 >> 32);
    g1[6] = (int)(unsigned)q3;  g1[7] = (int)(unsigned)(q3 >> 32);

    i32x4 gz = {0, 0, 0, 0};
#if defined(__clang_major__) && __clang_major__ >= 23
    i32x8 gz8 = {0, 0, 0, 0, 0, 0, 0, 0};
    __builtin_amdgcn_tensor_load_to_lds(g0, g1, gz, gz, gz8, 0);
#else
    __builtin_amdgcn_tensor_load_to_lds(g0, g1, gz, gz, 0);
#endif
}
#endif  // HAVE_TDM

// ---------------------------------------------------------------------------
// Generic bf16 WMMA GEMM:  C[M,N] = scale * op(A)[M,K] @ op(B)[K,N] + epilogue
//   transA: A element (m,k) at A[k*lda + m]   (else A[m*lda + k])
//   transB: B element (k,n) at B[n*ldb + k]   (else B[k*ldb + n])
//   tdmA  : stage A via TDM (requires !transA);  tdmB: requires transB
//   bias  : +bias[n]; maskN: +(1-maskN[n])*-1e30; maskM: *maskM[m] (nullable)
//   outF  : f32 output (nullable); outB: bf16 output (nullable)
// Block: 256 threads = 8 waves (4 in M x 2 in N); wave tile 32x64.
// ---------------------------------------------------------------------------
__global__ __launch_bounds__(256)
void gemm_bf16_wmma(const __bf16* __restrict__ A, long long strideA, int lda, int transA,
                    const __bf16* __restrict__ B, long long strideB, int ldb, int transB,
                    int tdmA, int tdmB,
                    int M, int N, int K, float scale,
                    const float* __restrict__ bias,
                    const float* __restrict__ maskN,
                    const float* __restrict__ maskM, int maskStride,
                    float* __restrict__ outF, long long strideOF,
                    __bf16* __restrict__ outB, long long strideOB, int ldc)
{
    __shared__ __bf16 lA[BM][LDSA];   // [m][k]
    __shared__ __bf16 lB[BN][LDSB];   // [n][k] (k-contiguous per column)

    const int tid  = threadIdx.x;
    const int lane = tid & 31;
    const int wid  = tid >> 5;
    const int wm   = wid & 3;     // wave row 0..3  -> 32 rows each
    const int wn   = wid >> 2;    // wave col 0..1  -> 64 cols each
    const int l15  = lane & 15;
    const int l5   = lane >> 4;   // half-wave select

    const int bz = blockIdx.z;
    const int m0 = blockIdx.y * BM;
    const int n0 = blockIdx.x * BN;

    const __bf16* Ab = A + (long long)bz * strideA;
    const __bf16* Bb = B + (long long)bz * strideB;

#if HAVE_TDM
    const int uA = tdmA;
    const int uB = tdmB;
    const unsigned ldsA_addr = (unsigned)(unsigned long long)(void*)&lA[0][0];
    const unsigned ldsB_addr = (unsigned)(unsigned long long)(void*)&lB[0][0];
#else
    const int uA = 0;
    const int uB = 0;
    (void)tdmA; (void)tdmB;
#endif

    v8f acc[2][4];
    const v8f vzero = {0.f, 0.f, 0.f, 0.f, 0.f, 0.f, 0.f, 0.f};
#pragma unroll
    for (int i = 0; i < 2; ++i)
#pragma unroll
        for (int j = 0; j < 4; ++j) acc[i][j] = vzero;

    for (int k0 = 0; k0 < K; k0 += BK) {
#if HAVE_TDM
        // ---- TDM staging: wave 0 only issues the DMA descriptors ----
        if (wid == 0) {
            if (uA)
                tdm_load_tile_bf16(ldsA_addr, &Ab[(long long)m0 * lda + k0],
                                   BK, BM, lda);
            if (uB)
                tdm_load_tile_bf16(ldsB_addr, &Bb[(long long)n0 * ldb + k0],
                                   BK, BN, ldb);
        }
#endif
        // ---- manual staging for layouts the TDM cannot produce ----
        if (!uA) {
            if (!transA) {
                for (int i = tid; i < (BM * BK) / 8; i += 256) {
                    int r = i >> 2;
                    int c = (i & 3) * 8;
                    *(v8bf*)&lA[r][c] =
                        *(const v8bf*)&Ab[(long long)(m0 + r) * lda + k0 + c];
                }
            } else {
                // A(m,k) = Ab[k*lda + m] : coalesced over m
                for (int i = tid; i < BM * BK; i += 256) {
                    int kk = i >> 7;
                    int mm = i & 127;
                    lA[mm][kk] = Ab[(long long)(k0 + kk) * lda + m0 + mm];
                }
            }
        }
        if (!uB) {
            if (!transB) {
                // B(k,n) = Bb[k*ldb + n] : coalesced over n, transpose to LDS
                for (int i = tid; i < BN * BK; i += 256) {
                    int kk = i >> 7;
                    int nn = i & 127;
                    lB[nn][kk] = Bb[(long long)(k0 + kk) * ldb + n0 + nn];
                }
            } else {
                for (int i = tid; i < (BN * BK) / 8; i += 256) {
                    int nn = i >> 2;
                    int c  = (i & 3) * 8;
                    *(v8bf*)&lB[nn][c] =
                        *(const v8bf*)&Bb[(long long)(n0 + nn) * ldb + k0 + c];
                }
            }
        }
#if HAVE_TDM
        if ((uA | uB) && wid == 0) __builtin_amdgcn_s_wait_tensorcnt(0);
#endif
        __syncthreads();

        // ---- fragments per the 16-bit WMMA lane layouts ----
        // A 16x32: lanes 0-15 = rows; elems 0..7 -> K=l5*8.., 8..15 -> K=16+l5*8..
        v16bf afrag[2];
#pragma unroll
        for (int i = 0; i < 2; ++i) {
            int mrow = wm * 32 + i * 16 + l15;
            union { v16bf v; v8bf h[2]; } u;
            u.h[0] = *(const v8bf*)&lA[mrow][l5 * 8];
            u.h[1] = *(const v8bf*)&lA[mrow][16 + l5 * 8];
            afrag[i] = u.v;
        }
        // B 32x16: col=l15; elems 0..7 -> K=l5*8.., 8..15 -> K=16+l5*8..
        v16bf bfrag[4];
#pragma unroll
        for (int j = 0; j < 4; ++j) {
            int ncol = wn * 64 + j * 16 + l15;
            union { v16bf v; v8bf h[2]; } u;
            u.h[0] = *(const v8bf*)&lB[ncol][l5 * 8];
            u.h[1] = *(const v8bf*)&lB[ncol][16 + l5 * 8];
            bfrag[j] = u.v;
        }

#pragma unroll
        for (int i = 0; i < 2; ++i)
#pragma unroll
            for (int j = 0; j < 4; ++j)
                acc[i][j] = __builtin_amdgcn_wmma_f32_16x16x32_bf16(
                    false, afrag[i], false, bfrag[j],
                    (short)0, acc[i][j], false, false);
        __syncthreads();
    }

    // ---- epilogue ----
    const float* mN = maskN ? maskN + (long long)bz * maskStride : nullptr;
    const float* mM = maskM ? maskM + (long long)bz * maskStride : nullptr;
    float*  oF = outF ? outF + (long long)bz * strideOF : nullptr;
    __bf16* oB = outB ? outB + (long long)bz * strideOB : nullptr;

#pragma unroll
    for (int i = 0; i < 2; ++i) {
#pragma unroll
        for (int j = 0; j < 4; ++j) {
            const int n = n0 + wn * 64 + j * 16 + l15;
            const float bi = bias ? bias[n] : 0.f;
            const float mb = mN ? (1.f - mN[n]) * (-1e30f) : 0.f;
#pragma unroll
            for (int r = 0; r < 8; ++r) {
                const int m = m0 + wm * 32 + i * 16 + l5 * 8 + r;
                float v = acc[i][j][r] * scale + bi + mb;
                if (mM) v *= mM[m];
                const long long idx = (long long)m * ldc + n;
                if (oF) oF[idx] = v;
                if (oB) oB[idx] = (__bf16)v;
            }
        }
    }
}

// ---------------------------------------------------------------------------
// f32 -> bf16 conversion
// ---------------------------------------------------------------------------
__global__ void f32_to_bf16_kernel(const float* __restrict__ in,
                                   __bf16* __restrict__ out, long long n)
{
    long long i = (long long)blockIdx.x * blockDim.x + threadIdx.x;
    const long long stride = (long long)gridDim.x * blockDim.x;
    for (; i < n; i += stride) out[i] = (__bf16)in[i];
}

// ---------------------------------------------------------------------------
// Row softmax over L2: one block per row, 1 read + 1 bf16 write.
// ---------------------------------------------------------------------------
__global__ __launch_bounds__(256)
void row_softmax_kernel(const float* __restrict__ a, __bf16* __restrict__ a2, int L2)
{
    const float* row = a  + (long long)blockIdx.x * L2;
    __bf16*     orow = a2 + (long long)blockIdx.x * L2;
    const int t = threadIdx.x;

    float v[8];
    float mx = -INFINITY;
#pragma unroll
    for (int i = 0; i < 8; ++i) {
        v[i] = row[t + 256 * i];
        mx = fmaxf(mx, v[i]);
    }
    __shared__ float red[256];
    red[t] = mx; __syncthreads();
    for (int s = 128; s > 0; s >>= 1) {
        if (t < s) red[t] = fmaxf(red[t], red[t + s]);
        __syncthreads();
    }
    mx = red[0]; __syncthreads();

    float sum = 0.f;
#pragma unroll
    for (int i = 0; i < 8; ++i) { v[i] = __expf(v[i] - mx); sum += v[i]; }
    red[t] = sum; __syncthreads();
    for (int s = 128; s > 0; s >>= 1) {
        if (t < s) red[t] += red[t + s];
        __syncthreads();
    }
    const float inv = 1.f / red[0];
#pragma unroll
    for (int i = 0; i < 8; ++i) orow[t + 256 * i] = (__bf16)(v[i] * inv);
}

// ---------------------------------------------------------------------------
// Column softmax over L1 (axis=1). Thread-per-column online softmax; columns
// are the fast axis so every pass is fully coalesced. Output stays in [l,m]
// layout; the consuming GEMM uses transA.
// ---------------------------------------------------------------------------
__global__ __launch_bounds__(256)
void col_softmax_kernel(const float* __restrict__ a, __bf16* __restrict__ a1,
                        int L1, int L2)
{
    const float* ab = a  + (long long)blockIdx.y * L1 * L2;
    __bf16*      ob = a1 + (long long)blockIdx.y * L1 * L2;
    const int m = blockIdx.x * 256 + threadIdx.x;

    float mx = -INFINITY, sum = 0.f;
    for (int l = 0; l < L1; ++l) {
        const float x = ab[(long long)l * L2 + m];
        const float nm = fmaxf(mx, x);
        sum = sum * __expf(mx - nm) + __expf(x - nm);
        mx = nm;
    }
    const float inv = 1.f / sum;
    for (int l = 0; l < L1; ++l) {
        const float x = ab[(long long)l * L2 + m];
        ob[(long long)l * L2 + m] = (__bf16)(__expf(x - mx) * inv);
    }
}

// ---------------------------------------------------------------------------
// Host side
// ---------------------------------------------------------------------------
static void run_gemm(hipStream_t stream,
                     const __bf16* A, long long sA, int lda, int tA,
                     const __bf16* Bp, long long sB, int ldb, int tB,
                     int tdmA, int tdmB,
                     int M, int N, int K, int batch, float scale,
                     const float* bias, const float* maskN, const float* maskM,
                     int maskStride,
                     float* oF, long long sOF, __bf16* oB, long long sOB, int ldc)
{
    dim3 grid(N / BN, M / BM, batch);
    gemm_bf16_wmma<<<grid, dim3(256), 0, stream>>>(
        A, sA, lda, tA, Bp, sB, ldb, tB, tdmA, tdmB, M, N, K, scale,
        bias, maskN, maskM, maskStride, oF, sOF, oB, sOB, ldc);
}

extern "C" void kernel_launch(void* const* d_in, const int* in_sizes, int n_in,
                              void* d_out, int out_size, void* d_ws, size_t ws_size,
                              hipStream_t stream)
{
    constexpr int BATCH = 8, L1 = 2048, L2 = 2048, D = 1024;
    constexpr long long nX = (long long)BATCH * L1 * D;   // 16,777,216
    constexpr long long nW = (long long)D * D;

    const float* x1   = (const float*)d_in[0];
    const float* x2   = (const float*)d_in[1];
    const float* mask = (const float*)d_in[2];   // [B, L2]
    const float* Wq   = (const float*)d_in[3];
    const float* bq   = (const float*)d_in[4];
    const float* Wk   = (const float*)d_in[5];
    const float* bk   = (const float*)d_in[6];
    const float* Wv   = (const float*)d_in[7];
    const float* bv   = (const float*)d_in[8];

    float* x1_out = (float*)d_out;                 // [B,L1,D]
    float* x2_out = (float*)d_out + nX;            // [B,L2,D]

    // ---- workspace layout (aliased; peak ~384MB) ----
    // 0..128MB   : q, k, v1, v2 (bf16, 32MB each)
    // 128..200MB : Wq/Wk/Wv bf16 (6MB) + x1_bf/x2_bf (64MB) -- dead after
    //              projections; region reused for the f32 score tensor
    // 128..256MB : scores a (f32, 128MB)
    // 256..320MB : a2 (bf16), 320..384MB : a1 (bf16)
    // mid (32MB) reuses q's slot (dead after scores).
    char* ws = (char*)d_ws;
    const size_t MB = 1ull << 20;
    __bf16* q_bf   = (__bf16*)(ws +   0 * MB);
    __bf16* k_bf   = (__bf16*)(ws +  32 * MB);
    __bf16* v1_bf  = (__bf16*)(ws +  64 * MB);
    __bf16* v2_bf  = (__bf16*)(ws +  96 * MB);
    __bf16* wq_bf  = (__bf16*)(ws + 128 * MB);
    __bf16* wk_bf  = (__bf16*)(ws + 130 * MB);
    __bf16* wv_bf  = (__bf16*)(ws + 132 * MB);
    __bf16* x1_bf  = (__bf16*)(ws + 136 * MB);
    __bf16* x2_bf  = (__bf16*)(ws + 168 * MB);
    float*  a_ws   = (float*) (ws + 128 * MB);
    __bf16* a2_bf  = (__bf16*)(ws + 256 * MB);
    __bf16* a1_bf  = (__bf16*)(ws + 320 * MB);
    __bf16* mid_bf = q_bf;

    const float scale = 0.03125f;  // 1/sqrt(1024)

    // 1) f32 -> bf16 staging
    f32_to_bf16_kernel<<<2048, 256, 0, stream>>>(x1, x1_bf, nX);
    f32_to_bf16_kernel<<<2048, 256, 0, stream>>>(x2, x2_bf, nX);
    f32_to_bf16_kernel<<<1024, 256, 0, stream>>>(Wq, wq_bf, nW);
    f32_to_bf16_kernel<<<1024, 256, 0, stream>>>(Wk, wk_bf, nW);
    f32_to_bf16_kernel<<<1024, 256, 0, stream>>>(Wv, wv_bf, nW);

    // 2) projections (batch folded into M): y = x @ W^T + b -> bf16 (TDM A+B)
    run_gemm(stream, x1_bf, 0, D, 0, wq_bf, 0, D, 1, 1, 1,
             BATCH * L1, D, D, 1, 1.f,
             bq, nullptr, nullptr, 0, nullptr, 0, q_bf, 0, D);
    run_gemm(stream, x2_bf, 0, D, 0, wk_bf, 0, D, 1, 1, 1,
             BATCH * L2, D, D, 1, 1.f,
             bk, nullptr, nullptr, 0, nullptr, 0, k_bf, 0, D);
    run_gemm(stream, x1_bf, 0, D, 0, wv_bf, 0, D, 1, 1, 1,
             BATCH * L1, D, D, 1, 1.f,
             bv, nullptr, nullptr, 0, nullptr, 0, v1_bf, 0, D);
    run_gemm(stream, x2_bf, 0, D, 0, wv_bf, 0, D, 1, 1, 1,
             BATCH * L2, D, D, 1, 1.f,
             bv, nullptr, nullptr, 0, nullptr, 0, v2_bf, 0, D);

    // 3) scores: a = scale * q @ k^T + (1-mask[n])*NEG -> f32 (TDM A+B)
    run_gemm(stream, q_bf, (long long)L1 * D, D, 0,
             k_bf, (long long)L2 * D, D, 1, 1, 1,
             L1, L2, D, BATCH, scale,
             nullptr, mask, nullptr, L2,
             a_ws, (long long)L1 * L2, nullptr, 0, L2);

    // 4) softmaxes: a2 over L2 (rows), a1 over L1 (columns)
    row_softmax_kernel<<<BATCH * L1, 256, 0, stream>>>(a_ws, a2_bf, L2);
    col_softmax_kernel<<<dim3(L2 / 256, BATCH), 256, 0, stream>>>(a_ws, a1_bf, L1, L2);

    // 5) x2_out = a2 @ v2 -> f32 output (TDM A)
    run_gemm(stream, a2_bf, (long long)L1 * L2, L2, 0,
             v2_bf, (long long)L2 * D, D, 0, 1, 0,
             L1, D, L2, BATCH, 1.f,
             nullptr, nullptr, nullptr, 0,
             x2_out, (long long)L1 * D, nullptr, 0, D);

    // 6) x1_mid = (a1^T @ v1) * mask[m] -> bf16 [B,L2,D] (manual staging)
    run_gemm(stream, a1_bf, (long long)L1 * L2, L2, 1,
             v1_bf, (long long)L1 * D, D, 0, 0, 0,
             L2, D, L1, BATCH, 1.f,
             nullptr, nullptr, mask, L2,
             nullptr, 0, mid_bf, (long long)L2 * D, D);

    // 7) x1_out = a2 @ x1_mid -> f32 output (TDM A)
    run_gemm(stream, a2_bf, (long long)L1 * L2, L2, 0,
             mid_bf, (long long)L2 * D, D, 0, 1, 0,
             L1, D, L2, BATCH, 1.f,
             nullptr, nullptr, nullptr, 0,
             x1_out, (long long)L1 * D, nullptr, 0, D);
}
